// MultiHeadAttention_80161269613128
// MI455X (gfx1250) — matrix operand order repaired and checked by reading, not compiled, not run on hardware
//
#include <hip/hip_runtime.h>
#include <hip/hip_bf16.h>

typedef __attribute__((ext_vector_type(16))) _Float16 v16h;
typedef __attribute__((ext_vector_type(8)))  float    v8f;
typedef unsigned int u32;

#define BQ 4
#define SQ 2048
#define DQ 512
#define HQ 8
#define DKQ 64

union FragH {
    uint4 u4[2];
    u32   u[8];
    v16h  h;
};

__device__ __forceinline__ u32 pack2f16(float x, float y) {
    union { _Float16 h[2]; u32 u; } p;
    p.h[0] = (_Float16)x; p.h[1] = (_Float16)y;
    return p.u;
}

__device__ __forceinline__ v8f wmma_f16(const v16h& a, const v16h& b, const v8f& c) {
    return __builtin_amdgcn_wmma_f32_16x16x32_f16(false, a, false, b, (short)0, c, false, false);
}

// ---------------------------------------------------------------------------
// GEMM: out[m,n] = sum_k X[m,k] * W[n,k] + bias[n]     (torch Linear x@W.T+b)
// M = B*S = 8192, N = K = 512.
// MODE 0: Q -> f16 [B,H,S,DK], scaled by 1/sqrt(DK)
// MODE 1: K -> f16 [B,H,S,DK]
// MODE 2: V -> f16 [B,H,DK,S] (transposed for PV B-fragments)
// MODE 3: O -> fp32 [B,S,D], input X is f16 (context)
// Block: 256 threads (8 waves). Block tile 128x64. Wave tile 32x32. K-step 32.
// ---------------------------------------------------------------------------
template<int MODE>
__global__ void __launch_bounds__(256)
mha_gemm_kernel(const void* Xv, const float* __restrict__ W,
                const float* __restrict__ bias, void* Outv) {
    __shared__ u32 Alds[128 * 20];   // 128 rows x 32 f16, padded pitch 20 u32
    __shared__ u32 Blds[64 * 20];    // 64 rows x 32 f16

    const int tid  = threadIdx.x;
    const int lane = tid & 31;
    const int wave = tid >> 5;
    const int lr   = lane & 15;
    const int hlf  = lane >> 4;
    const int wm   = wave >> 1;      // 0..3 : M sub-tile
    const int wn   = wave & 1;       // 0..1 : N sub-tile
    const int m0   = blockIdx.x * 128;
    const int n0   = blockIdx.y * 64;

    v8f acc[2][2];
    #pragma unroll
    for (int i = 0; i < 2; ++i)
        #pragma unroll
        for (int j = 0; j < 2; ++j) acc[i][j] = (v8f){};

    const float* Xf = (const float*)Xv;
    const u32*   Xu = (const u32*)Xv;     // f16 view for MODE 3

    for (int kk = 0; kk < DQ; kk += 32) {
        // ---- stage A tile (128x32) into LDS as f16 ----
        if constexpr (MODE == 3) {
            #pragma unroll
            for (int i = 0; i < 4; ++i) {
                int flat = tid + 256 * i;
                int row = flat >> 3, seg = flat & 7;
                uint2 d = *(const uint2*)&Xu[(size_t)(m0 + row) * 256 + (kk >> 1) + seg * 2];
                Alds[row * 20 + seg * 2]     = d.x;
                Alds[row * 20 + seg * 2 + 1] = d.y;
            }
        } else {
            #pragma unroll
            for (int i = 0; i < 4; ++i) {
                int flat = tid + 256 * i;
                int row = flat >> 3, seg = flat & 7;
                float4 d = *(const float4*)&Xf[(size_t)(m0 + row) * DQ + kk + seg * 4];
                Alds[row * 20 + seg * 2]     = pack2f16(d.x, d.y);
                Alds[row * 20 + seg * 2 + 1] = pack2f16(d.z, d.w);
            }
        }
        // ---- stage B tile (64 rows of W, 32 k) ----
        #pragma unroll
        for (int i = 0; i < 2; ++i) {
            int flat = tid + 256 * i;
            int row = flat >> 3, seg = flat & 7;
            float4 d = *(const float4*)&W[(size_t)(n0 + row) * DQ + kk + seg * 4];
            Blds[row * 20 + seg * 2]     = pack2f16(d.x, d.y);
            Blds[row * 20 + seg * 2 + 1] = pack2f16(d.z, d.w);
        }
        __syncthreads();

        // ---- fragments per ISA 7.12.2 layouts ----
        FragH a[2], b[2];
        #pragma unroll
        for (int mi = 0; mi < 2; ++mi) {
            const u32* ar = &Alds[(wm * 32 + mi * 16 + lr) * 20];
            a[mi].u4[0] = *(const uint4*)&ar[4 * hlf];        // K 8h..8h+7
            a[mi].u4[1] = *(const uint4*)&ar[8 + 4 * hlf];    // K 16+8h..16+8h+7
        }
        #pragma unroll
        for (int ni = 0; ni < 2; ++ni) {
            const u32* br = &Blds[(wn * 32 + ni * 16 + lr) * 20];
            b[ni].u4[0] = *(const uint4*)&br[4 * hlf];        // K 8h..8h+7
            b[ni].u4[1] = *(const uint4*)&br[8 + 4 * hlf];    // K 16+8h..16+8h+7
        }
        #pragma unroll
        for (int mi = 0; mi < 2; ++mi)
            #pragma unroll
            for (int ni = 0; ni < 2; ++ni)
                acc[mi][ni] = wmma_f16(a[mi].h, b[ni].h, acc[mi][ni]);
        __syncthreads();
    }

    // ---- epilogue: bias + layout-specific store ----
    #pragma unroll
    for (int ni = 0; ni < 2; ++ni) {
        const int ng = n0 + wn * 32 + ni * 16 + lr;
        const float bv = bias[ng];
        #pragma unroll
        for (int mi = 0; mi < 2; ++mi) {
            const int mbase = m0 + wm * 32 + mi * 16 + 8 * hlf;
            v8f c = acc[mi][ni];
            if constexpr (MODE == 0 || MODE == 1) {
                _Float16* dst = (_Float16*)Outv;
                const int hh = ng >> 6, dk = ng & 63;
                #pragma unroll
                for (int r = 0; r < 8; ++r) {
                    int mg = mbase + r;
                    int bb = mg >> 11, s = mg & 2047;
                    float v = c[r] + bv;
                    if constexpr (MODE == 0) v *= 0.125f;   // 1/sqrt(64)
                    dst[(((size_t)bb * HQ + hh) * SQ + s) * DKQ + dk] = (_Float16)v;
                }
            } else if constexpr (MODE == 2) {
                const int hh = ng >> 6, dk = ng & 63;
                const int bb = mbase >> 11, s0 = mbase & 2047;
                union { _Float16 h8[8]; uint4 q; } pk;
                #pragma unroll
                for (int r = 0; r < 8; ++r) pk.h8[r] = (_Float16)(c[r] + bv);
                _Float16* dst = (_Float16*)Outv +
                                (((size_t)bb * HQ + hh) * DKQ + dk) * SQ + s0;
                *(uint4*)dst = pk.q;   // 8 consecutive s, 16B aligned
            } else {
                float* dst = (float*)Outv;
                #pragma unroll
                for (int r = 0; r < 8; ++r) {
                    int mg = mbase + r;
                    dst[(size_t)mg * DQ + ng] = c[r] + bv;
                }
            }
        }
    }
}

// ---------------------------------------------------------------------------
// Flash attention (causal), scale pre-folded into Q.
// Block = 256 threads = 8 waves; wave owns 16 query rows; block covers 128.
// grid = (S/128, B*H)
// ---------------------------------------------------------------------------
__global__ void __launch_bounds__(256)
mha_attn_kernel(const _Float16* __restrict__ Qf, const _Float16* __restrict__ Kf,
                const _Float16* __restrict__ Vt, _Float16* __restrict__ Ctx) {
    __shared__ u32 Pl[8 * 16 * 36];   // per-wave 16x64 f16 P tile, pitch 36 u32

    const int tid  = threadIdx.x;
    const int lane = tid & 31;
    const int wave = tid >> 5;
    const int lr   = lane & 15;
    const int hlf  = lane >> 4;
    const int bh   = blockIdx.y;                 // 0..B*H-1
    const int q0   = blockIdx.x * 128 + wave * 16;

    // Q A-fragments (16x64 -> two 16x32 chunks), resident in registers
    FragH aq[2];
    {
        const u32* qrow = (const u32*)(Qf + ((size_t)bh * SQ + q0 + lr) * DKQ);
        #pragma unroll
        for (int c = 0; c < 2; ++c) {
            aq[c].u4[0] = *(const uint4*)&qrow[c * 16 + 4 * hlf];
            aq[c].u4[1] = *(const uint4*)&qrow[c * 16 + 8 + 4 * hlf];
        }
    }

    v8f acc[4];
    #pragma unroll
    for (int i = 0; i < 4; ++i) acc[i] = (v8f){};
    float mrow[8], lrow[8];
    #pragma unroll
    for (int r = 0; r < 8; ++r) { mrow[r] = -3.0e38f; lrow[r] = 0.0f; }

    _Float16*  pw = (_Float16*)&Pl[wave * 576];
    const u32* pr = &Pl[wave * 576 + lr * 36];

    for (int jb = 0; jb <= q0 + 15; jb += 64) {
        // ---- scores S = Q * K^T  (16 x 64) ----
        v8f sc[4];
        #pragma unroll
        for (int nj = 0; nj < 4; ++nj) {
            v8f s = (v8f){};
            #pragma unroll
            for (int c = 0; c < 2; ++c) {
                const u32* kp = (const u32*)(Kf +
                    ((size_t)bh * SQ + jb + nj * 16 + lr) * DKQ + c * 32 + hlf * 8);
                FragH bk;
                bk.u4[0] = *(const uint4*)kp;
                bk.u4[1] = *(const uint4*)(kp + 8);
                s = wmma_f16(aq[c].h, bk.h, s);
            }
            sc[nj] = s;
        }

        // ---- causal mask on diagonal blocks ----
        if (jb + 63 > q0) {
            #pragma unroll
            for (int nj = 0; nj < 4; ++nj) {
                const int j = jb + nj * 16 + lr;
                #pragma unroll
                for (int r = 0; r < 8; ++r) {
                    const int q = q0 + r + 8 * hlf;
                    if (j > q) sc[nj][r] = -1.0e30f;
                }
            }
        }

        // ---- online softmax (row reductions across 16-lane halves) ----
        float mnew[8], alpha[8];
        #pragma unroll
        for (int r = 0; r < 8; ++r) {
            float t = fmaxf(fmaxf(sc[0][r], sc[1][r]), fmaxf(sc[2][r], sc[3][r]));
            t = fmaxf(t, __shfl_xor(t, 8, 32));
            t = fmaxf(t, __shfl_xor(t, 4, 32));
            t = fmaxf(t, __shfl_xor(t, 2, 32));
            t = fmaxf(t, __shfl_xor(t, 1, 32));
            float mn = fmaxf(mrow[r], t);
            mnew[r]  = mn;
            alpha[r] = __expf(mrow[r] - mn);
            mrow[r]  = mn;
        }
        #pragma unroll
        for (int nj = 0; nj < 4; ++nj)
            #pragma unroll
            for (int r = 0; r < 8; ++r)
                sc[nj][r] = __expf(sc[nj][r] - mnew[r]);
        #pragma unroll
        for (int r = 0; r < 8; ++r) {
            float t = sc[0][r] + sc[1][r] + sc[2][r] + sc[3][r];
            t += __shfl_xor(t, 8, 32);
            t += __shfl_xor(t, 4, 32);
            t += __shfl_xor(t, 2, 32);
            t += __shfl_xor(t, 1, 32);
            lrow[r] = lrow[r] * alpha[r] + t;
            acc[0][r] *= alpha[r]; acc[1][r] *= alpha[r];
            acc[2][r] *= alpha[r]; acc[3][r] *= alpha[r];
        }

        // ---- P: C-layout -> A-layout via per-wave LDS ----
        #pragma unroll
        for (int nj = 0; nj < 4; ++nj)
            #pragma unroll
            for (int r = 0; r < 8; ++r)
                pw[(r + 8 * hlf) * 72 + nj * 16 + lr] = (_Float16)sc[nj][r];
        asm volatile("s_wait_dscnt 0" ::: "memory");

        // ---- ctx += P * V ----
        #pragma unroll
        for (int c = 0; c < 2; ++c) {
            FragH ap;
            ap.u4[0] = *(const uint4*)&pr[c * 16 + 4 * hlf];
            ap.u4[1] = *(const uint4*)&pr[c * 16 + 8 + 4 * hlf];
            #pragma unroll
            for (int nd = 0; nd < 4; ++nd) {
                const u32* vp = (const u32*)(Vt +
                    ((size_t)bh * DKQ + nd * 16 + lr) * SQ + jb + c * 32 + hlf * 8);
                FragH bv;
                bv.u4[0] = *(const uint4*)vp;
                bv.u4[1] = *(const uint4*)(vp + 8);
                acc[nd] = wmma_f16(ap.h, bv.h, acc[nd]);
            }
        }
        asm volatile("s_wait_dscnt 0" ::: "memory");
    }

    // ---- normalize + store context f16 [B,S,D] ----
    #pragma unroll
    for (int r = 0; r < 8; ++r) lrow[r] = 1.0f / lrow[r];
    const int bb = bh >> 3, hh = bh & 7;
    #pragma unroll
    for (int nd = 0; nd < 4; ++nd)
        #pragma unroll
        for (int r = 0; r < 8; ++r) {
            const int s = q0 + r + 8 * hlf;
            const float v = acc[nd][r] * lrow[r];
            Ctx[((size_t)bb * SQ + s) * DQ + hh * DKQ + nd * 16 + lr] = (_Float16)v;
        }
}

extern "C" void kernel_launch(void* const* d_in, const int* in_sizes, int n_in,
                              void* d_out, int out_size, void* d_ws, size_t ws_size,
                              hipStream_t stream) {
    const float* q   = (const float*)d_in[0];
    const float* k   = (const float*)d_in[1];
    const float* v   = (const float*)d_in[2];
    // d_in[3]: causal mask (hardcoded in the attention kernel)
    const float* w_q = (const float*)d_in[4];
    const float* b_q = (const float*)d_in[5];
    const float* w_k = (const float*)d_in[6];
    const float* b_k = (const float*)d_in[7];
    const float* w_v = (const float*)d_in[8];
    const float* b_v = (const float*)d_in[9];
    const float* w_o = (const float*)d_in[10];
    const float* b_o = (const float*)d_in[11];

    char* ws = (char*)d_ws;
    const size_t SEG = (size_t)BQ * HQ * SQ * DKQ * sizeof(_Float16);   // 8 MB
    _Float16* Qf = (_Float16*)(ws);
    _Float16* Kf = (_Float16*)(ws + SEG);
    _Float16* Vt = (_Float16*)(ws + 2 * SEG);
    _Float16* Cx = (_Float16*)(ws + 3 * SEG);

    const dim3 blk(256);
    const dim3 gg(BQ * SQ / 128, DQ / 64);          // 64 x 8
    mha_gemm_kernel<0><<<gg, blk, 0, stream>>>(q, w_q, b_q, Qf);
    mha_gemm_kernel<1><<<gg, blk, 0, stream>>>(k, w_k, b_k, Kf);
    mha_gemm_kernel<2><<<gg, blk, 0, stream>>>(v, w_v, b_v, Vt);

    const dim3 ga(SQ / 128, BQ * HQ);               // 16 x 32
    mha_attn_kernel<<<ga, blk, 0, stream>>>(Qf, Kf, Vt, Cx);

    mha_gemm_kernel<3><<<gg, blk, 0, stream>>>(Cx, w_o, b_o, d_out);
}
